// LlamaTPAttention_49804440765058
// MI455X (gfx1250) — compile-verified, matrix operand order checked by reading, not run on hardware
//
#include <hip/hip_runtime.h>
#include <cstdint>
#include <cstddef>

#define S_LEN 1024
#define DIM   8192
#define HEADS 128
#define HDIM  64

// POD vector types (HIP's uint4/float4 have non-trivial ctors -> unusable in unions)
typedef unsigned int u32x4 __attribute__((ext_vector_type(4)));
typedef unsigned int u32x2 __attribute__((ext_vector_type(2)));
typedef float        f32x4 __attribute__((ext_vector_type(4)));

typedef __attribute__((ext_vector_type(16))) __bf16 v16bf;
typedef __attribute__((ext_vector_type(8)))  float  v8f;

union Frag {
    u32x4 q[2];
    v16bf v;
};

__device__ __forceinline__ unsigned short f2bf(float f) {
    unsigned int u = __float_as_uint(f);
    u += 0x7FFFu + ((u >> 16) & 1u);   // round-to-nearest-even
    return (unsigned short)(u >> 16);
}

// ---- CDNA5 async global->LDS DMA (ASYNCcnt-tracked, no VGPR data) ----------
// LDS byte offset = low 32 bits of the generic pointer (flat-LDS aperture
// truncates to addr[31:0], which is what async VDST expects).
__device__ __forceinline__ unsigned int lds_off(const void* p) {
    return (unsigned int)(uintptr_t)p;
}
__device__ __forceinline__ void async_copy_b128(unsigned int dst_lds, const void* src) {
    asm volatile("global_load_async_to_lds_b128 %0, %1, off"
                 :: "v"(dst_lds), "v"((unsigned long long)(uintptr_t)src)
                 : "memory");
}
__device__ __forceinline__ void wait_async_all() {
    asm volatile("s_wait_asynccnt 0x0" ::: "memory");
}

// cross-lane reductions within groups of 16 lanes (wave32): xor 1,2,4,8 stay in-half
__device__ __forceinline__ float row_max16(float v) {
    v = fmaxf(v, __int_as_float(__builtin_amdgcn_ds_swizzle(__float_as_int(v), 0x041F)));
    v = fmaxf(v, __int_as_float(__builtin_amdgcn_ds_swizzle(__float_as_int(v), 0x081F)));
    v = fmaxf(v, __int_as_float(__builtin_amdgcn_ds_swizzle(__float_as_int(v), 0x101F)));
    v = fmaxf(v, __int_as_float(__builtin_amdgcn_ds_swizzle(__float_as_int(v), 0x201F)));
    return v;
}
__device__ __forceinline__ float row_sum16(float v) {
    v += __int_as_float(__builtin_amdgcn_ds_swizzle(__float_as_int(v), 0x041F));
    v += __int_as_float(__builtin_amdgcn_ds_swizzle(__float_as_int(v), 0x081F));
    v += __int_as_float(__builtin_amdgcn_ds_swizzle(__float_as_int(v), 0x101F));
    v += __int_as_float(__builtin_amdgcn_ds_swizzle(__float_as_int(v), 0x201F));
    return v;
}

// ---------------------------------------------------------------------------
// fp32 -> bf16 conversion (x activation), b128 in / b128 out
// ---------------------------------------------------------------------------
__global__ __launch_bounds__(256) void cvt_kernel(const float* __restrict__ x,
                                                  unsigned short* __restrict__ y,
                                                  int n) {
    int i = (blockIdx.x * 256 + threadIdx.x) * 8;
    if (i >= n) return;
    f32x4 a = *(const f32x4*)(x + i);
    f32x4 b = *(const f32x4*)(x + i + 4);
    u32x4 out;
    out.x = (unsigned)f2bf(a.x) | ((unsigned)f2bf(a.y) << 16);
    out.y = (unsigned)f2bf(a.z) | ((unsigned)f2bf(a.w) << 16);
    out.z = (unsigned)f2bf(b.x) | ((unsigned)f2bf(b.y) << 16);
    out.w = (unsigned)f2bf(b.z) | ((unsigned)f2bf(b.w) << 16);
    *(u32x4*)(y + i) = out;
}

// ---------------------------------------------------------------------------
// C[M,N] = A_bf16[M,K] @ W_f32[K,N].  Double-buffered software pipeline:
//   - A tile (bf16):  global_load_async_to_lds_b128 (ASYNCcnt DMA)
//   - W tile (fp32):  global f32x4 -> in-register bf16 convert -> ds_store_b64
// 128x128 block, K-step 32, 8 waves (2M x 4N), 4x2 wmma accs per wave.
// LDS row stride 40 bf16 (80 B): aligned b128 fragment loads, conflict-free.
// ---------------------------------------------------------------------------
template <typename OutT>
__global__ __launch_bounds__(256) void gemm_xw_kernel(
    const unsigned short* __restrict__ A,   // [M][K] bf16
    const float*          __restrict__ W,   // [K][N] f32
    OutT*                 __restrict__ C,   // [M][N]
    int M, int N, int K)
{
    __shared__ unsigned short a_lds[2][128 * 40];
    __shared__ unsigned short b_lds[2][128 * 40];   // transposed: [n][k]

    const int t    = threadIdx.x;
    const int wave = t >> 5;
    const int lane = t & 31;
    const int l16  = lane & 15;
    const int lh   = lane >> 4;
    const int wm   = wave & 1;    // wave M index -> 64 rows
    const int wn   = wave >> 1;   // wave N index -> 32 cols
    const int m0   = blockIdx.y * 128;
    const int n0   = blockIdx.x * 128;

    // A-stage coords: 2 threads per row, 16 bf16 each (2x b128 DMA)
    const int arow = t >> 1;
    const int acol = (t & 1) * 16;
    // B-stage coords: each thread owns a 4(K) x 4(N) sub-block
    const int bk = (t >> 5) * 4;     // 0..28
    const int bn = (t & 31) * 4;     // 0..124

    v8f acc[4][2];
#pragma unroll
    for (int i = 0; i < 4; ++i)
#pragma unroll
        for (int j = 0; j < 2; ++j)
            for (int e = 0; e < 8; ++e) acc[i][j][e] = 0.0f;

    f32x4 wreg[4];

    auto issue_a = [&](int k0, int buf) {
        const unsigned short* src = A + (size_t)(m0 + arow) * K + k0 + acol;
        unsigned int dst = lds_off(&a_lds[buf][arow * 40 + acol]);
        async_copy_b128(dst,      src);
        async_copy_b128(dst + 16, src + 8);
    };
    auto load_b_global = [&](int k0) {
#pragma unroll
        for (int j = 0; j < 4; ++j) {
            const float* wp = W + (size_t)(k0 + bk + j) * N + n0 + bn;
            wreg[j] = *(const f32x4*)wp;
        }
        if (k0 + 32 < K)
            __builtin_prefetch(W + (size_t)(k0 + 32 + bk) * N + n0 + bn, 0, 1);
    };
    auto store_b_lds = [&](int buf) {
#pragma unroll
        for (int c = 0; c < 4; ++c) {   // pack 4 K-values per column -> b64 store
            u32x2 v;
            v.x = (unsigned)f2bf(wreg[0][c]) | ((unsigned)f2bf(wreg[1][c]) << 16);
            v.y = (unsigned)f2bf(wreg[2][c]) | ((unsigned)f2bf(wreg[3][c]) << 16);
            *(u32x2*)(&b_lds[buf][(bn + c) * 40 + bk]) = v;
        }
    };

    // prologue: stage tile 0 into buffer 0
    issue_a(0, 0);
    load_b_global(0);
    store_b_lds(0);

    const int nk = K / 32;
    for (int kt = 0; kt < nk; ++kt) {
        const int cur = kt & 1;
        const int nxt = cur ^ 1;

        wait_async_all();     // this wave's async A-tile DMA done
        __syncthreads();      // all waves' A DMA + B stores visible

        if (kt + 1 < nk) {    // kick off next tile while we compute
            issue_a((kt + 1) * 32, nxt);
            load_b_global((kt + 1) * 32);
        }

        Frag af[4], bf[2];
#pragma unroll
        for (int i = 0; i < 4; ++i) {
            int off = (wm * 64 + i * 16 + l16) * 40 + lh * 8;
            af[i].q[0] = *(const u32x4*)(&a_lds[cur][off]);
            af[i].q[1] = *(const u32x4*)(&a_lds[cur][off + 16]);
        }
#pragma unroll
        for (int j = 0; j < 2; ++j) {
            int off = (wn * 32 + j * 16 + l16) * 40 + lh * 8;
            bf[j].q[0] = *(const u32x4*)(&b_lds[cur][off]);
            bf[j].q[1] = *(const u32x4*)(&b_lds[cur][off + 16]);
        }
#pragma unroll
        for (int i = 0; i < 4; ++i)
#pragma unroll
            for (int j = 0; j < 2; ++j)
                acc[i][j] = __builtin_amdgcn_wmma_f32_16x16x32_bf16(
                    false, af[i].v, false, bf[j].v, (short)0, acc[i][j], false, false);

        if (kt + 1 < nk) store_b_lds(nxt);
        // barrier at top of next iteration closes the pipeline stage
    }

    // epilogue: C/D layout -> lane l16 = N, VGPR r = M row (+8 for upper half)
#pragma unroll
    for (int i = 0; i < 4; ++i)
#pragma unroll
        for (int j = 0; j < 2; ++j)
#pragma unroll
            for (int r = 0; r < 8; ++r) {
                int m = m0 + wm * 64 + i * 16 + lh * 8 + r;
                int n = n0 + wn * 32 + j * 16 + l16;
                float v = acc[i][j][r];
                if constexpr (sizeof(OutT) == 2)
                    ((unsigned short*)C)[(size_t)m * N + n] = f2bf(v);
                else
                    ((float*)C)[(size_t)m * N + n] = v;
            }
}

// ---------------------------------------------------------------------------
// Flash attention: block = (head, 128-query tile), 8 waves x 16 rows.
// Q and K tiles staged via async global->LDS DMA; V transposed through VGPRs.
// S=QK^T via WMMA, online softmax in fp32 (ds_swizzle row reductions),
// P restaged per-wave in LDS to WMMA A-layout, O += P@V via WMMA.
// ---------------------------------------------------------------------------
__global__ __launch_bounds__(256) void attn_kernel(
    const unsigned short* __restrict__ Q,
    const unsigned short* __restrict__ Kb,
    const unsigned short* __restrict__ V,
    unsigned short* __restrict__ O)
{
    __shared__ unsigned short q_lds[128 * 72];
    __shared__ unsigned short k_lds[64 * 72];
    __shared__ unsigned short vT_lds[64 * 72];      // [d][key]
    __shared__ unsigned short p_lds[8 * 16 * 72];   // per-wave P tile

    const int t    = threadIdx.x;
    const int wave = t >> 5;
    const int lane = t & 31;
    const int l16  = lane & 15;
    const int lh   = lane >> 4;
    const int h    = blockIdx.x >> 3;
    const int q0   = (blockIdx.x & 7) * 128;
    const size_t hoff = (size_t)h * HDIM;

    // Q tile 128x64 via async DMA (drained by first in-loop wait)
    {
        int row  = t >> 1;
        int half = (t & 1) * 32;
        const unsigned short* src = Q + (size_t)(q0 + row) * DIM + hoff + half;
        unsigned int dst = lds_off(&q_lds[row * 72 + half]);
#pragma unroll
        for (int i = 0; i < 4; ++i)
            async_copy_b128(dst + i * 16, src + i * 8);
    }

    v8f   o_acc[4];
    float m_acc[8], l_acc[8];
#pragma unroll
    for (int e = 0; e < 8; ++e) { m_acc[e] = -3.0e38f; l_acc[e] = 0.0f; }
#pragma unroll
    for (int dt = 0; dt < 4; ++dt)
        for (int e = 0; e < 8; ++e) o_acc[dt][e] = 0.0f;

    unsigned short* p_wave = p_lds + wave * 16 * 72;
    const float scale = 0.125f;   // 1/sqrt(HDIM)

    for (int kb0 = 0; kb0 < S_LEN; kb0 += 64) {
        __syncthreads();             // previous K/V tiles fully consumed
        {
            int row = t >> 2;        // key 0..63
            int d0  = (t & 3) * 16;
            // K tile: straight bf16 copy -> async DMA
            const unsigned short* ks = Kb + (size_t)(kb0 + row) * DIM + hoff + d0;
            unsigned int kd = lds_off(&k_lds[row * 72 + d0]);
            async_copy_b128(kd,      ks);
            async_copy_b128(kd + 16, ks + 8);
            // V tile: needs [key][d] -> [d][key] transpose, through VGPRs
            const u32x4* vs = (const u32x4*)(V + (size_t)(kb0 + row) * DIM + hoff + d0);
            u32x4 va = vs[0], vb2 = vs[1];
            unsigned int u[8] = {va.x, va.y, va.z, va.w, vb2.x, vb2.y, vb2.z, vb2.w};
#pragma unroll
            for (int j = 0; j < 8; ++j) {
                int d = d0 + 2 * j;
                vT_lds[(size_t)d * 72 + row]       = (unsigned short)(u[j] & 0xFFFFu);
                vT_lds[(size_t)(d + 1) * 72 + row] = (unsigned short)(u[j] >> 16);
            }
        }
        wait_async_all();            // Q (first iter) + K DMA landed
        __syncthreads();

        // S = Q K^T : 16 rows x 64 keys per wave
        v8f s_c[4];
#pragma unroll
        for (int n = 0; n < 4; ++n)
            for (int e = 0; e < 8; ++e) s_c[n][e] = 0.0f;
#pragma unroll
        for (int s = 0; s < 2; ++s) {
            Frag aq;
            int aoff = (wave * 16 + l16) * 72 + s * 32 + lh * 8;
            aq.q[0] = *(const u32x4*)(q_lds + aoff);
            aq.q[1] = *(const u32x4*)(q_lds + aoff + 16);
#pragma unroll
            for (int n = 0; n < 4; ++n) {
                Frag bk;
                int boff = (n * 16 + l16) * 72 + s * 32 + lh * 8;
                bk.q[0] = *(const u32x4*)(k_lds + boff);
                bk.q[1] = *(const u32x4*)(k_lds + boff + 16);
                s_c[n] = __builtin_amdgcn_wmma_f32_16x16x32_bf16(
                    false, aq.v, false, bk.v, (short)0, s_c[n], false, false);
            }
        }

        // online softmax (fp32)
        float alpha[8];
#pragma unroll
        for (int r = 0; r < 8; ++r) {
            float mr = -3.0e38f;
#pragma unroll
            for (int n = 0; n < 4; ++n) {
                float sv = s_c[n][r] * scale;
                s_c[n][r] = sv;
                mr = fmaxf(mr, sv);
            }
            mr = row_max16(mr);
            float m_new = fmaxf(m_acc[r], mr);
            alpha[r] = __expf(m_acc[r] - m_new);
            float rs = 0.0f;
#pragma unroll
            for (int n = 0; n < 4; ++n) {
                float p = __expf(s_c[n][r] - m_new);
                rs += p;
                p_wave[(lh * 8 + r) * 72 + n * 16 + l16] = f2bf(p);
            }
            rs = row_sum16(rs);
            l_acc[r] = l_acc[r] * alpha[r] + rs;
            m_acc[r] = m_new;
        }
#pragma unroll
        for (int dt = 0; dt < 4; ++dt)
#pragma unroll
            for (int r = 0; r < 8; ++r) o_acc[dt][r] *= alpha[r];

        // O += P @ V  (P in per-wave LDS A-layout; V^T gives contiguous K)
#pragma unroll
        for (int s = 0; s < 2; ++s) {
            Frag ap;
            int aoff = l16 * 72 + s * 32 + lh * 8;
            ap.q[0] = *(const u32x4*)(p_wave + aoff);
            ap.q[1] = *(const u32x4*)(p_wave + aoff + 16);
#pragma unroll
            for (int dt = 0; dt < 4; ++dt) {
                Frag bv;
                int boff = (dt * 16 + l16) * 72 + s * 32 + lh * 8;
                bv.q[0] = *(const u32x4*)(vT_lds + boff);
                bv.q[1] = *(const u32x4*)(vT_lds + boff + 16);
                o_acc[dt] = __builtin_amdgcn_wmma_f32_16x16x32_bf16(
                    false, ap.v, false, bv.v, (short)0, o_acc[dt], false, false);
            }
        }
    }

    // normalize and store (bf16)
#pragma unroll
    for (int r = 0; r < 8; ++r) {
        float inv = 1.0f / l_acc[r];
        int m = q0 + wave * 16 + lh * 8 + r;
#pragma unroll
        for (int dt = 0; dt < 4; ++dt)
            O[(size_t)m * DIM + hoff + dt * 16 + l16] = f2bf(o_acc[dt][r] * inv);
    }
}

// ---------------------------------------------------------------------------
extern "C" void kernel_launch(void* const* d_in, const int* in_sizes, int n_in,
                              void* d_out, int out_size, void* d_ws, size_t ws_size,
                              hipStream_t stream) {
    (void)in_sizes; (void)n_in; (void)out_size; (void)ws_size;
    const float* x  = (const float*)d_in[0];
    const float* wq = (const float*)d_in[1];
    const float* wk = (const float*)d_in[2];
    const float* wv = (const float*)d_in[3];
    const float* wo = (const float*)d_in[4];

    const size_t SD = (size_t)S_LEN * DIM;           // 8M elements
    unsigned short* xb = (unsigned short*)d_ws;      // bf16 buffers in workspace
    unsigned short* qb = xb + SD;
    unsigned short* kb = qb + SD;
    unsigned short* vb = kb + SD;
    unsigned short* ob = vb + SD;                    // total 80 MB of ws

    cvt_kernel<<<(int)(SD / (256 * 8)), 256, 0, stream>>>(x, xb, (int)SD);

    dim3 gg(DIM / 128, S_LEN / 128);                 // 64 x 8 blocks
    gemm_xw_kernel<unsigned short><<<gg, 256, 0, stream>>>(xb, wq, qb, S_LEN, DIM, DIM);
    gemm_xw_kernel<unsigned short><<<gg, 256, 0, stream>>>(xb, wk, kb, S_LEN, DIM, DIM);
    gemm_xw_kernel<unsigned short><<<gg, 256, 0, stream>>>(xb, wv, vb, S_LEN, DIM, DIM);

    attn_kernel<<<HEADS * (S_LEN / 128), 256, 0, stream>>>(qb, kb, vb, ob);

    gemm_xw_kernel<float><<<gg, 256, 0, stream>>>(ob, wo, (float*)d_out, S_LEN, DIM, DIM);
}
